// LSTMLayer_67413806678549
// MI455X (gfx1250) — matrix operand order repaired and checked by reading, not compiled, not run on hardware
//
#include <hip/hip_runtime.h>
#include <hip/hip_bf16.h>
#include <stdint.h>

// CDNA5 wave32 WMMA / descriptor types
typedef __attribute__((ext_vector_type(16))) __bf16 v16bf;
typedef __attribute__((ext_vector_type(8)))  float  v8f;
typedef __attribute__((ext_vector_type(4)))  int    v4i;
typedef __attribute__((ext_vector_type(8)))  int    v8i;

#define T_STEPS 2048
#define BATCH   64
#define IDIM    256
#define HDIM    256
#define GDIM    1024   // 4*HDIM
#define HSTRIDE 264    // padded bf16 row stride (528B = 33*16B, 16B aligned, bank-spread)
#define XP_SLOTS (64 * 32 * 8)   // bf16 elems of one (t, m_tile) x_proj block = 32KB

union FragAB { v16bf v; uint4 q[2]; };
union FragO  { uint4 q; __bf16 h[8]; };

__device__ __forceinline__ float fast_sigmoid(float x) {
  return 1.0f / (1.0f + __expf(-x));
}
__device__ __forceinline__ float fast_tanh(float x) {
  return 2.0f / (1.0f + __expf(-2.0f * x)) - 1.0f;
}

// Issue one TDM 1-D copy: 32KB contiguous global -> LDS (4096 x 8B elements).
// D# group0/group1 built per CDNA5 ISA 08_async_tensor (§8.3/8.4). Uniform operands
// go in SGPRs; tracked with TENSORcnt.
__device__ __forceinline__ void tdm_load_32k(const __bf16* gsrc, __bf16* lds_dst) {
  uint64_t ga = (uint64_t)(uintptr_t)gsrc;
  uint32_t la = (uint32_t)(uintptr_t)lds_dst;   // flat low 32 bits == LDS offset
  v4i g0;
  g0[0] = 1;                                // count=1, user descriptor
  g0[1] = (int)la;                          // lds_addr
  g0[2] = (int)(uint32_t)ga;                // global_addr[31:0]
  g0[3] = (int)(((uint32_t)(ga >> 32) & 0x01FFFFFFu) | 0x80000000u); // addr[56:32]|type=2
  v8i g1;
  g1[0] = 0x00030000;                       // workgroup_mask=0, data_size=3 (8B)
  g1[1] = 0x10000000;                       // tensor_dim0 = 4096 (bits[63:48])
  g1[2] = 0x00010000;                       // tensor_dim1 = 1    (bits[95:80])
  g1[3] = 0x10000000;                       // tile_dim0   = 4096 (bits[127:112])
  g1[4] = 0x00000001;                       // tile_dim1   = 1
  g1[5] = 4096;                             // tensor_dim0_stride = 4096
  g1[6] = 0;
  g1[7] = 0;
  asm volatile("tensor_load_to_lds %0, %1" :: "s"(g0), "s"(g1) : "memory");
}

// ---------------- Phase 0a: fp32 -> bf16 conversion (8 elems / thread) ---------------
__global__ void cvt_f32_to_bf16_x8(const float* __restrict__ src,
                                   __bf16* __restrict__ dst, int n8) {
  int i = blockIdx.x * blockDim.x + threadIdx.x;
  if (i < n8) {
    float4 a = ((const float4*)src)[2 * i];
    float4 b = ((const float4*)src)[2 * i + 1];
    FragO o;
    o.h[0] = (__bf16)a.x; o.h[1] = (__bf16)a.y;
    o.h[2] = (__bf16)a.z; o.h[3] = (__bf16)a.w;
    o.h[4] = (__bf16)b.x; o.h[5] = (__bf16)b.y;
    o.h[6] = (__bf16)b.z; o.h[7] = (__bf16)b.w;
    ((uint4*)dst)[i] = o.q;
  }
}

// ---------------- Phase 0b: repack weights into per-lane B-fragment order ------------
// dst layout: [tile (64)][kchunk (8)][lane (32)][16 bf16]  -> every B fragment load is
// part of one contiguous 1KB burst per (tile, kchunk).
__global__ void repack_weight_frags(const float* __restrict__ w,
                                    __bf16* __restrict__ dst) {
  int idx = blockIdx.x * blockDim.x + threadIdx.x;   // < 64*8*32 = 16384
  int lane  = idx & 31;
  int kc    = (idx >> 5) & 7;
  int ntile = idx >> 8;
  int g  = ntile * 16 + (lane & 15);
  int k0 = kc * 32 + ((lane >> 4) * 8);
  const float* s = w + (long)g * 256 + k0;
  FragO o0, o1;
#pragma unroll
  for (int j = 0; j < 8; ++j) { o0.h[j] = (__bf16)s[j]; o1.h[j] = (__bf16)s[16 + j]; }
  uint4* d = (uint4*)(dst + (long)idx * 16);
  d[0] = o0.q;
  d[1] = o1.q;
}

__global__ void bias_sum_kernel(const float* __restrict__ a,
                                const float* __restrict__ b,
                                float* __restrict__ c) {
  int i = blockIdx.x * blockDim.x + threadIdx.x;
  if (i < GDIM) c[i] = a[i] + b[i];
}

// ---------------- Phase 1: x_proj = x @ W_ih^T + (b_ih + b_hh) ----------------
// Grid: T*4 blocks (block = t*4 + m_tile), 256 threads (8 waves). Wave w computes
// column tiles {w, w+8, ..., w+56}; A fragments hoisted into 64 VGPRs. Output stored
// as raw per-lane C fragments (bf16) consumed by phase 2 / TDM as contiguous 32KB.
__global__ void __launch_bounds__(256)
lstm_xproj_kernel(const __bf16* __restrict__ xb,
                  const __bf16* __restrict__ wfrag,
                  const float*  __restrict__ bias,
                  __bf16* __restrict__ xproj) {
  const int lane = threadIdx.x & 31;
  const int wave = threadIdx.x >> 5;
  const int blk  = blockIdx.x;          // t*4 + m_tile
  const int col  = lane & 15;
  const int kh8  = (lane >> 4) * 8;     // A-fragment K sub-offset

  // Hoist all A fragments (16x256 slice of x, bf16) into registers.
  const __bf16* ab = xb + ((long)blk * 16 + col) * IDIM;
  FragAB A[8];
#pragma unroll
  for (int kc = 0; kc < 8; ++kc) {
    A[kc].q[0] = *(const uint4*)(ab + kc * 32 + kh8);
    A[kc].q[1] = *(const uint4*)(ab + kc * 32 + 16 + kh8);
  }

  for (int it = 0; it < 8; ++it) {
    const int ntile = wave + it * 8;
    v8f acc = {};
#pragma unroll
    for (int kc = 0; kc < 8; ++kc) {
      FragAB B;
      const uint4* bp = (const uint4*)(wfrag + ((long)(ntile * 8 + kc) * 32 + lane) * 16);
      B.q[0] = bp[0];
      B.q[1] = bp[1];
      acc = __builtin_amdgcn_wmma_f32_16x16x32_bf16(
          false, A[kc].v, false, B.v, (short)0, acc, false, false);
    }
    const float bv = bias[ntile * 16 + col];
    FragO o;
#pragma unroll
    for (int i = 0; i < 8; ++i) o.h[i] = (__bf16)(acc[i] + bv);
    *(uint4*)(xproj + (((long)blk * 64 + ntile) * 32 + lane) * 8) = o.q;
  }
}

// ---------------- Phase 2: sequential scan (persistent, 4 WGs, 16 batch rows each) ----
// 16 waves; wave w owns gate tiles {w, w+16, w+32, w+48} == (i,f,g,o) for h-columns
// [16w, 16w+16): the whole cell update is wave-local, c stays in one v8f fragment.
// h is shared via a double-buffered bf16 LDS tile (A-fragment layout, padded rows).
// x_proj fragments for step t+1 are prefetched into LDS by the Tensor Data Mover
// while step t computes; wave 0 waits on TENSORcnt before the per-step barrier.
__global__ void __launch_bounds__(512, 1)
lstm_recurrent_kernel(const __bf16* __restrict__ xproj,
                      const __bf16* __restrict__ wfrag,
                      float* __restrict__ out) {
  __shared__ __bf16 hbuf[2][16][HSTRIDE];
  __shared__ __align__(16) __bf16 xpstage[2][XP_SLOTS];

  const int lane = threadIdx.x & 31;
  const int wave = threadIdx.x >> 5;
  const int mt   = blockIdx.x;          // batch slice 0..3
  const int col  = lane & 15;
  const int kh8  = (lane >> 4) * 8;

  // h(0) = 0
  for (int i = threadIdx.x; i < 2 * 16 * HSTRIDE; i += 512)
    ((__bf16*)hbuf)[i] = (__bf16)0.0f;

  // Prefetch x_proj block for t=0.
  if (wave == 0) {
    tdm_load_32k(xproj + (long)mt * XP_SLOTS, &xpstage[0][0]);
    __builtin_amdgcn_s_wait_tensorcnt(0);
  }
  __syncthreads();

  v8f cf = {};        // cell state fragment (f32, register resident for all 2048 steps)
  float hv[8];        // last-step h values in f32 for the final store

#pragma clang loop unroll(disable)
  for (int t = 0; t < T_STEPS; ++t) {
    const int rbuf = t & 1;
    const int wbuf = rbuf ^ 1;

    // Kick off TDM prefetch of step t+1's x_proj block (overlaps this step's GEMM).
    if (wave == 0 && t + 1 < T_STEPS) {
      tdm_load_32k(xproj + ((long)(t + 1) * 4 + mt) * XP_SLOTS, &xpstage[wbuf][0]);
    }

    // Seed accumulators with staged x-projection fragments (bias already folded in).
    v8f acc[4];
#pragma unroll
    for (int g = 0; g < 4; ++g) {
      FragO xf;
      xf.q = ((const uint4*)&xpstage[rbuf][0])[(wave + 16 * g) * 32 + lane];
      v8f a;
#pragma unroll
      for (int i = 0; i < 8; ++i) a[i] = (float)xf.h[i];
      acc[g] = a;
    }

    // gates += h @ W_hh^T   (K = 256, 8 chunks of 32; A from LDS, B fragments from L2)
#pragma unroll
    for (int kc = 0; kc < 8; ++kc) {
      FragAB A;
      const __bf16* ap = &hbuf[rbuf][col][kc * 32 + kh8];
      A.q[0] = *(const uint4*)(ap);
      A.q[1] = *(const uint4*)(ap + 16);
#pragma unroll
      for (int g = 0; g < 4; ++g) {
        FragAB B;
        const uint4* bp =
            (const uint4*)(wfrag + ((long)((wave + 16 * g) * 8 + kc) * 32 + lane) * 16);
        B.q[0] = bp[0];
        B.q[1] = bp[1];
        acc[g] = __builtin_amdgcn_wmma_f32_16x16x32_bf16(
            false, A.v, false, B.v, (short)0, acc[g], false, false);
      }
    }

    // Wave-local LSTM cell update: acc[0]=i, acc[1]=f, acc[2]=g, acc[3]=o.
#pragma unroll
    for (int i = 0; i < 8; ++i) {
      const float ig = fast_sigmoid(acc[0][i]);
      const float fg = fast_sigmoid(acc[1][i]);
      const float gg = fast_tanh(acc[2][i]);
      const float og = fast_sigmoid(acc[3][i]);
      const float c  = fg * cf[i] + ig * gg;
      cf[i] = c;
      const float h = og * fast_tanh(c);
      hv[i] = h;
      const int m = i + 8 * (lane >> 4);       // batch row within slice (C layout)
      hbuf[wbuf][m][wave * 16 + col] = (__bf16)h;
    }

    // Ensure the prefetched block has landed before anyone reads it next step.
    if (wave == 0) __builtin_amdgcn_s_wait_tensorcnt(0);
    __syncthreads();
  }

  // Final h (f32, last-step values) -> d_out [B, H]
#pragma unroll
  for (int i = 0; i < 8; ++i) {
    const int m = i + 8 * (lane >> 4);
    out[(mt * 16 + m) * HDIM + wave * 16 + col] = hv[i];
  }
}

// ---------------- Launcher ----------------
extern "C" void kernel_launch(void* const* d_in, const int* in_sizes, int n_in,
                              void* d_out, int out_size, void* d_ws, size_t ws_size,
                              hipStream_t stream) {
  const float* x    = (const float*)d_in[0];   // [T, B, I]
  const float* W_ih = (const float*)d_in[1];   // [4H, I]
  const float* W_hh = (const float*)d_in[2];   // [4H, H]
  const float* b_ih = (const float*)d_in[3];   // [4H]
  const float* b_hh = (const float*)d_in[4];   // [4H]

  char* ws = (char*)d_ws;
  __bf16* xb      = (__bf16*)(ws);                    // 2048*64*256 bf16 = 67,108,864 B
  __bf16* wihfrag = (__bf16*)(ws + 67108864);         // 1024*256 bf16   =    524,288 B
  __bf16* whhfrag = (__bf16*)(ws + 67633152);         // 1024*256 bf16   =    524,288 B
  float*  bias    = (float*) (ws + 68157440);         // 1024 f32        =      4,096 B
  __bf16* xproj   = (__bf16*)(ws + 68161536);         // frags, 268,435,456 B

  const int nx8 = (T_STEPS * BATCH * IDIM) / 8;       // 4,194,304

  cvt_f32_to_bf16_x8<<<(nx8 + 255) / 256, 256, 0, stream>>>(x, xb, nx8);
  repack_weight_frags<<<64, 256, 0, stream>>>(W_ih, wihfrag);
  repack_weight_frags<<<64, 256, 0, stream>>>(W_hh, whhfrag);
  bias_sum_kernel<<<4, 256, 0, stream>>>(b_ih, b_hh, bias);

  lstm_xproj_kernel<<<T_STEPS * 4, 256, 0, stream>>>(xb, wihfrag, bias, xproj);
  lstm_recurrent_kernel<<<4, 512, 0, stream>>>(xproj, whhfrag, (float*)d_out);
}
